// MultiDescriptorGMVAE_25374666785267
// MI455X (gfx1250) — compile-verified, matrix operand order checked by reading, not run on hardware
//
#include <hip/hip_runtime.h>
#include <hip/hip_bf16.h>

typedef __attribute__((ext_vector_type(16))) __bf16 v16bf;
typedef __attribute__((ext_vector_type(8)))  __bf16 v8bf;
typedef __attribute__((ext_vector_type(8)))  float  v8f;
typedef __attribute__((ext_vector_type(4)))  unsigned int v4u;
typedef __attribute__((ext_vector_type(8)))  unsigned int v8u;

#define B_SZ 16384
#define C_SZ 4096
#define D_SZ 256
#define K_SZ 512       // 2*D: [z^2 | -2z] x [inv_var | mu*inv_var]
#define KSTEP 64       // K macro-step staged via TDM
#define LDS_STRIDE 72  // 64 elems + 8 pad elems (TDM pad: 16B per 128B row) -> conflict-free

// ---------------------------------------------------------------------------
// Pack A = [z^2, -2z] as bf16, row-major [B, 512]
// ---------------------------------------------------------------------------
__global__ __launch_bounds__(256) void gmvae_pack_z(const float* __restrict__ z,
                                                    __bf16* __restrict__ a) {
  const int b = blockIdx.x;
  const int d = threadIdx.x;
  float v = z[(size_t)b * D_SZ + d];
  a[(size_t)b * K_SZ + d]        = (__bf16)(v * v);
  a[(size_t)b * K_SZ + D_SZ + d] = (__bf16)(-2.0f * v);
}

// ---------------------------------------------------------------------------
// Pack W = [inv_var, mu*inv_var] as bf16, row-major [C, 512];
// bias[c] = sum_d(mu^2*inv_var + logvar)
// ---------------------------------------------------------------------------
__global__ __launch_bounds__(256) void gmvae_pack_w(const float* __restrict__ mu,
                                                    const float* __restrict__ logvar,
                                                    __bf16* __restrict__ w,
                                                    float* __restrict__ bias) {
  __shared__ float red[256];
  const int c = blockIdx.x;
  const int d = threadIdx.x;
  float lv = logvar[(size_t)c * D_SZ + d];
  float m  = mu[(size_t)c * D_SZ + d];
  float iv = __expf(-lv);
  w[(size_t)c * K_SZ + d]        = (__bf16)iv;
  w[(size_t)c * K_SZ + D_SZ + d] = (__bf16)(m * iv);
  red[d] = m * m * iv + lv;
  __syncthreads();
  for (int s = 128; s > 0; s >>= 1) {
    if (d < s) red[d] += red[d + s];
    __syncthreads();
  }
  if (d == 0) bias[c] = red[0];
}

// ---------------------------------------------------------------------------
// TDM tile load: 128 rows x 64 bf16 cols from a [rows,512] bf16 tensor into
// LDS at ldsAddr, with 16B padding inserted after every 128B row
// (row stride in LDS = 144B = LDS_STRIDE elems).
//   D# group0: count=1 | lds_addr | global_addr | type=2
//   D# group1: data_size=2B, pad_enable, pad_interval=32dw, pad_amount=4dw,
//              tensor_dim0=512, tensor_dim1=rows, tile=64x128, dim0_stride=512
// ---------------------------------------------------------------------------
__device__ __forceinline__ void tdm_load_tile(unsigned ldsAddr,
                                              const __bf16* gptr,
                                              unsigned rows) {
  unsigned long long ga = (unsigned long long)(uintptr_t)gptr;
  v4u g0;
  g0[0] = 1u;                                      // count=1, user descriptor
  g0[1] = ldsAddr;                                 // LDS byte address
  g0[2] = (unsigned)ga;                            // global_addr[31:0]
  g0[3] = (unsigned)(ga >> 32) | 0x80000000u;      // global_addr[56:32] | type=2

  v8u g1;
  g1[0] = (1u << 16)      // data_size = 1 -> 2 bytes
        | (1u << 20)      // pad_enable
        | (4u << 22)      // pad_interval: code 4 -> every 32 DWORDs (128B)
        | (3u << 25);     // pad_amount:   code 3 -> 4 DWORDs (16B)
  g1[1] = (512u & 0xffffu) << 16;                  // tensor_dim0[15:0] (=512)
  g1[2] = (rows & 0xffffu) << 16;                  // dim0 hi=0 | tensor_dim1 lo
  g1[3] = (rows >> 16) | (64u << 16);              // tensor_dim1 hi | tile_dim0=64
  g1[4] = 128u;                                    // tile_dim1=128 | tile_dim2=0
  g1[5] = 512u;                                    // tensor_dim0_stride lo (=512)
  g1[6] = 0u;                                      // stride hi | dim1_stride lo
  g1[7] = 0u;                                      // dim1_stride hi

  asm volatile("tensor_load_to_lds %0, %1" : : "s"(g0), "s"(g1) : "memory");
}

// ---------------------------------------------------------------------------
// Tiled bf16 WMMA GEMM with TDM double-buffered LDS staging.
// Workgroup tile: 128(M) x 128(N). 8 waves in a 2(M)x4(N) grid; each wave owns
// 4x2 = 8 accumulators of 16x16.  K loop: 8 macro-steps of 64 (2 WMMA K=32
// sub-steps each).  Wave 0 drives the TDM; all waves consume from LDS.
// ---------------------------------------------------------------------------
__global__ __launch_bounds__(256, 1)
void gmvae_gemm_tdm(const __bf16* __restrict__ A, const __bf16* __restrict__ W,
                    const float* __restrict__ bias, float* __restrict__ logits) {
  __shared__ __bf16 shA[2][128 * LDS_STRIDE];   // 2 x 18 KB
  __shared__ __bf16 shB[2][128 * LDS_STRIDE];   // 2 x 18 KB

  const int lane  = threadIdx.x & 31;
  const int wave  = threadIdx.x >> 5;
  const int waveM = wave & 1;   // 2 wave rows
  const int waveN = wave >> 1;  // 4 wave cols

  const int mBase = blockIdx.y * 128 + waveM * 64;  // 4 tiles of 16
  const int nBase = blockIdx.x * 128 + waveN * 32;  // 2 tiles of 16

  const int mLane = lane & 15;
  const int kSel  = (lane >> 4) << 3;  // lanes 0-15 -> K base 0, lanes 16-31 -> 8

  const __bf16* gA = A + (size_t)(blockIdx.y * 128) * K_SZ;  // A tile rows
  const __bf16* gW = W + (size_t)(blockIdx.x * 128) * K_SZ;  // W tile rows

  const bool producer = (threadIdx.x < 32);

  v8f acc[4][2] = {};
  union Frag { v16bf v; v8bf h[2]; };

  // Prologue: kick off buffer 0.
  if (producer) {
    tdm_load_tile((unsigned)(uintptr_t)&shA[0][0], gA, B_SZ);
    tdm_load_tile((unsigned)(uintptr_t)&shB[0][0], gW, C_SZ);
  }

  int buf = 0;
  for (int k = 0; k < K_SZ; k += KSTEP) {
    if (producer) {
      if (k + KSTEP < K_SZ) {
        tdm_load_tile((unsigned)(uintptr_t)&shA[buf ^ 1][0], gA + k + KSTEP, B_SZ);
        tdm_load_tile((unsigned)(uintptr_t)&shB[buf ^ 1][0], gW + k + KSTEP, C_SZ);
        __builtin_amdgcn_s_wait_tensorcnt(2);  // current buffer done (in-order)
      } else {
        __builtin_amdgcn_s_wait_tensorcnt(0);
      }
    }
    __syncthreads();  // LDS buffer `buf` ready for all waves

#pragma unroll
    for (int kk = 0; kk < KSTEP; kk += 32) {
      Frag af[4], bfm[2];
      // A fragments (16-bit 16x32 layout): two 16B chunks per lane.
#pragma unroll
      for (int i = 0; i < 4; ++i) {
        const int r = (waveM * 64 + i * 16 + mLane) * LDS_STRIDE + kk + kSel;
        af[i].h[0] = *(const v8bf*)&shA[buf][r];
        af[i].h[1] = *(const v8bf*)&shA[buf][r + 16];
      }
      // B fragments: column n of the 32x16 tile == W-tile row (contiguous).
#pragma unroll
      for (int j = 0; j < 2; ++j) {
        const int r = (waveN * 32 + j * 16 + mLane) * LDS_STRIDE + kk + kSel;
        bfm[j].h[0] = *(const v8bf*)&shB[buf][r];
        bfm[j].h[1] = *(const v8bf*)&shB[buf][r + 16];
      }
#pragma unroll
      for (int i = 0; i < 4; ++i)
#pragma unroll
        for (int j = 0; j < 2; ++j)
          acc[i][j] = __builtin_amdgcn_wmma_f32_16x16x32_bf16(
              false, af[i].v, false, bfm[j].v, (short)0, acc[i][j], false, false);
    }

    __syncthreads();  // all waves done with `buf` before TDM overwrites it
    buf ^= 1;
  }

  // Epilogue: C/D layout -> VGPR r, lanes 0-15: row r, col lane; lanes 16-31: row r+8
  const float cterm = -0.5f * 256.0f * 1.8378770664093453f   // -0.5*D*log(2pi)
                      - 8.317766166719343f;                  // log(1/4096)
  const int rBase = (lane >> 4) * 8;
#pragma unroll
  for (int j = 0; j < 2; ++j) {
    const int n    = nBase + j * 16 + mLane;
    const float bn = bias[n];
#pragma unroll
    for (int i = 0; i < 4; ++i) {
#pragma unroll
      for (int r = 0; r < 8; ++r) {
        const int m = mBase + i * 16 + rBase + r;
        logits[(size_t)m * C_SZ + n] = -0.5f * (acc[i][j][r] + bn) + cterm;
      }
    }
  }
}

// ---------------------------------------------------------------------------
// Row softmax + argmax: one block per row; full 4096-float row staged in LDS
// so logits are read once and probs written once.
// ---------------------------------------------------------------------------
__global__ __launch_bounds__(256)
void gmvae_softmax(const float* __restrict__ logits, float* __restrict__ probs,
                   float* __restrict__ pred) {
  __shared__ float row[C_SZ];
  __shared__ float red[256];
  __shared__ int   redi[256];
  const int b   = blockIdx.x;
  const int tid = threadIdx.x;
  const float* lrow = logits + (size_t)b * C_SZ;

  float lmax = -3.402823466e38f;
  int   lidx = 0;
  for (int j = tid; j < C_SZ; j += 256) {
    float v = lrow[j];
    row[j] = v;
    if (v > lmax) { lmax = v; lidx = j; }
  }
  red[tid] = lmax; redi[tid] = lidx;
  __syncthreads();
  for (int s = 128; s > 0; s >>= 1) {
    if (tid < s) {
      float o = red[tid + s]; int oi = redi[tid + s];
      if (o > red[tid] || (o == red[tid] && oi < redi[tid])) { red[tid] = o; redi[tid] = oi; }
    }
    __syncthreads();
  }
  const float m = red[0];
  const int amax = redi[0];
  __syncthreads();

  float ls = 0.0f;
  for (int j = tid; j < C_SZ; j += 256) {
    float e = __expf(row[j] - m);
    row[j] = e;
    ls += e;
  }
  red[tid] = ls;
  __syncthreads();
  for (int s = 128; s > 0; s >>= 1) {
    if (tid < s) red[tid] += red[tid + s];
    __syncthreads();
  }
  const float inv = 1.0f / red[0];
  float* prow = probs + (size_t)b * C_SZ;
  for (int j = tid; j < C_SZ; j += 256) prow[j] = row[j] * inv;
  if (tid == 0) pred[b] = (float)amax;
}

// ---------------------------------------------------------------------------
extern "C" void kernel_launch(void* const* d_in, const int* in_sizes, int n_in,
                              void* d_out, int out_size, void* d_ws, size_t ws_size,
                              hipStream_t stream) {
  const float* z      = (const float*)d_in[0];  // [B, D]
  const float* mu     = (const float*)d_in[1];  // [C, D]
  const float* logvar = (const float*)d_in[2];  // [C, D]

  float* out    = (float*)d_out;
  float* logits = out;                                   // [B, C]
  float* probs  = out + (size_t)B_SZ * C_SZ;             // [B, C]
  float* pred   = out + 2 * (size_t)B_SZ * C_SZ;         // [B]

  __bf16* aPack = (__bf16*)d_ws;                         // 16 MB
  __bf16* wPack = aPack + (size_t)B_SZ * K_SZ;           // 4 MB
  float*  bias  = (float*)(wPack + (size_t)C_SZ * K_SZ); // 16 KB

  gmvae_pack_z<<<B_SZ, 256, 0, stream>>>(z, aPack);
  gmvae_pack_w<<<C_SZ, 256, 0, stream>>>(mu, logvar, wPack, bias);

  dim3 grid(C_SZ / 128, B_SZ / 128);  // (32, 128)
  gmvae_gemm_tdm<<<grid, 256, 0, stream>>>(aPack, wPack, bias, logits);

  gmvae_softmax<<<B_SZ, 256, 0, stream>>>(logits, probs, pred);
}
